// MCVectorQuantizer_61984968016343
// MI455X (gfx1250) — compile-verified, matrix operand order checked by reading, not run on hardware
//
#include <hip/hip_runtime.h>
#include <hip/hip_bf16.h>
#include <cstdint>

// ---------------------------------------------------------------------------
// MCVectorQuantizer for MI455X (gfx1250, wave32, WMMA bf16 + async/TDM)
// ---------------------------------------------------------------------------

typedef __attribute__((ext_vector_type(16))) __bf16 v16bf;
typedef __attribute__((ext_vector_type(8)))  float  v8f;
typedef __attribute__((ext_vector_type(4)))  int    i32x4;
typedef __attribute__((ext_vector_type(8)))  int    i32x8;

#define TOKENS   8192      // B*T
#define NJ       32
#define ED       128
#define NE       1024
#define HIDN     256
#define TPB      128       // 4 waves
#define TOK_BLK  64

#define LSCALE   (1.25f / (8192.0f * 128.0f * 32.0f))   // (1+beta)/(B*T*C*V)

struct Level { int j[8]; int p[8]; int n; };

// Workspace layout (bytes)
#define WS_EMBF  0u                            // 32*64 tiles * 4096B bf16 frag tiles
#define WS_EMBF_SZ (32u*64u*4096u)             // 8,388,608
#define WS_W12   (WS_EMBF + WS_EMBF_SZ)        // w1f (131072B) + w2f (65536B)
#define WS_W12_SZ (196608u)
#define WS_EN    (WS_W12 + WS_W12_SZ)          // 32*1024 f32 norms
#define WS_EN_SZ (131072u)
#define WS_NEED  ((size_t)(WS_EN + WS_EN_SZ))

// ---- CDNA5 16-bit A/B fragment K mapping (ISA 7.12.2) ----
__device__ __forceinline__ int kmap16(int e, int half) {
  int g = e >> 1;
  int k = (g < 4) ? (2 * g + 8 * half) : (16 + 2 * (g - 4) + 8 * half);
  return k + (e & 1);
}
// inverse: element index within [ks][32 lanes][16 e] frag layout for (row m, K=k)
__device__ __forceinline__ int fragpos16(int m, int k) {
  int kk = k & 31, ks = k >> 5;
  int h  = (kk >> 3) & 1, hi = kk >> 4;
  int e  = 2 * (((kk & 7) >> 1) + 4 * hi) + (kk & 1);
  int ln = (m & 15) + 16 * h;
  return ((ks * 32 + ln) << 4) + e;
}
__device__ __forceinline__ unsigned pack_bf16(float a, float b) {
  unsigned short ua = __builtin_bit_cast(unsigned short, (__bf16)a);
  unsigned short ub = __builtin_bit_cast(unsigned short, (__bf16)b);
  return (unsigned)ua | ((unsigned)ub << 16);
}

// ---- CDNA5 async global->LDS (ASYNCcnt) ----
#define ASYNC_B128(ldsoff, gvoff, base)                                     \
  asm volatile("global_load_async_to_lds_b128 %0, %1, %2 offset:0"          \
               :: "v"(ldsoff), "v"(gvoff), "s"(base) : "memory")
#define WAIT_ASYNC0() asm volatile("s_wait_asynccnt 0" ::: "memory")

// ---- CDNA5 Tensor Data Mover: 1-D copy descriptor (data_size=8B) ----
__device__ __forceinline__ void tdm_load_1d(unsigned lds_off, const void* gbase,
                                            unsigned nbytes) {
  unsigned long long ga = (unsigned long long)(uintptr_t)gbase;
  unsigned n8 = nbytes >> 3;               // units of 8 bytes (<= 65535)
  i32x4 g0 = { (int)1u,                            // count=1, user D#
               (int)lds_off,                       // lds_addr
               (int)(unsigned)ga,                  // global_addr[31:0]
               (int)((unsigned)((ga >> 32) & 0x1FFFFFFu) | (2u << 30)) }; // type=2
  i32x8 g1 = { (int)(3u << 16),                    // data_size=8B, mask=0
               (int)((n8 & 0xFFFFu) << 16),        // tensor_dim0[15:0] @bit48
               (int)(1u << 16),                    // tensor_dim0[31:16]=0, tensor_dim1=1
               (int)((n8 & 0xFFFFu) << 16),        // tile_dim0 @bit112
               (int)1u,                            // tile_dim1=1, tile_dim2=0
               (int)n8,                            // tensor_dim0_stride[31:0]
               0, 0 };
  i32x4 gz4 = { 0, 0, 0, 0 };
  i32x8 gz8 = { 0, 0, 0, 0, 0, 0, 0, 0 };
  __builtin_amdgcn_tensor_load_to_lds(g0, g1, gz4, gz4, gz8, 0);
}
#define WAIT_TENSOR0() __builtin_amdgcn_s_wait_tensorcnt(0)

// ===========================================================================
// Prep kernels: one-time f32 -> bf16 swizzles into workspace
// ===========================================================================
__global__ void prep_w_kernel(const float* __restrict__ W1,
                              const float* __restrict__ W2,
                              __bf16* __restrict__ w12) {
  int f = blockIdx.x * 256 + threadIdx.x;
  if (f < 65536) {                       // W1: [8ks][16nt][32][16]
    int e = f & 15, ln = (f >> 4) & 31, nt = (f >> 9) & 15, ks = f >> 13;
    int k = ks * 32 + kmap16(e, ln >> 4);
    int n = nt * 16 + (ln & 15);
    w12[f] = (__bf16)W1[k * HIDN + n];
  } else if (f < 98304) {                // W2: [8ks][8nt][32][16]
    int f2 = f - 65536;
    int e = f2 & 15, ln = (f2 >> 4) & 31, nt = (f2 >> 9) & 7, ks = f2 >> 12;
    int k = ks * 32 + kmap16(e, ln >> 4);
    int n = nt * 16 + (ln & 15);
    w12[f] = (__bf16)W2[k * ED + n];
  }
}

__global__ void prep_emb_kernel(const float* __restrict__ emb,
                                __bf16* __restrict__ embf) {
  int t = blockIdx.x, j = blockIdx.y;    // 64 tiles x 32 joints
  const float* ej = emb + ((size_t)j * NE + t * 16) * ED;
  __bf16* o = embf + ((size_t)(j * 64 + t)) * 2048;
  for (int f = threadIdx.x; f < 2048; f += TPB) {   // [4ks][32][16]
    int e = f & 15, ln = (f >> 4) & 31, ks = f >> 9;
    int k = ks * 32 + kmap16(e, ln >> 4);
    o[f] = (__bf16)ej[(ln & 15) * ED + k];
  }
}

__global__ void prep_enorm_kernel(const float* __restrict__ emb,
                                  float* __restrict__ en) {
  int gid = blockIdx.x * TPB + threadIdx.x;   // 32*1024 codes
  const float* r = emb + (size_t)gid * ED;
  float s = 0.0f;
  #pragma unroll 8
  for (int k = 0; k < ED; ++k) s = fmaf(r[k], r[k], s);
  en[gid] = s;
}

__global__ void vq_init_kernel(float* loss) {
  if (threadIdx.x == 0 && blockIdx.x == 0) *loss = 0.0f;
}

// ===========================================================================
// Fast level kernel
// ===========================================================================
// LDS layout (bytes)
#define G_OFF_W1   0
#define G_OFF_W2   131072
#define G_OFF_FRX  (131072 + 65536)          // 4 waves * 8KB A-frag staging
#define G_OFF_STG  (G_OFF_FRX + 32768)       // 4 waves * [16][128] f32 (h2)
#define G_OFF_ET   (G_OFF_STG + 32768)       // 2 * 4KB bf16 codebook tiles
#define G_OFF_EN   (G_OFF_ET + 8192)         // 1024 f32 norms
#define G_OFF_B1   (G_OFF_EN + 4096)
#define G_OFF_LG   (G_OFF_B1 + 1024)
#define G_OFF_LB   (G_OFF_LG + 1024)
#define G_OFF_B2   (G_OFF_LB + 1024)
#define G_LDS_TOTAL (G_OFF_B2 + 512)         // 278016 B  (< 320KB/WGP)

__global__ __launch_bounds__(TPB, 1) void vq_level_fast(
    const float*  __restrict__ z,    const float* __restrict__ emb,
    const __bf16* __restrict__ embf, const __bf16* __restrict__ w12f,
    const float*  __restrict__ enormg,
    const float* __restrict__ b1,  const float* __restrict__ lng,
    const float* __restrict__ lnb, const float* __restrict__ b2,
    float* __restrict__ zq, float* __restrict__ loss, float* __restrict__ oidx,
    Level lv)
{
  extern __shared__ __attribute__((aligned(64))) char smem[];
  __bf16* w1l  = (__bf16*)(smem + G_OFF_W1);
  __bf16* w2l  = (__bf16*)(smem + G_OFF_W2);
  float*  enrm = (float*) (smem + G_OFF_EN);
  float*  b1s  = (float*) (smem + G_OFF_B1);
  float*  lgs  = (float*) (smem + G_OFF_LG);
  float*  lbs  = (float*) (smem + G_OFF_LB);
  float*  b2s  = (float*) (smem + G_OFF_B2);

  const int tid  = threadIdx.x;
  const int lane = tid & 31;
  const int wave = tid >> 5;
  const int half = lane >> 4;
  const int col  = lane & 15;
  const int jid  = lv.j[blockIdx.y];
  const int pid  = lv.p[blockIdx.y];
  const bool root = (pid < 0);

  const float*  embj  = emb  + (size_t)jid * NE * ED;
  const __bf16* embjt = embf + (size_t)jid * 64 * 2048;   // swizzled tiles
  __bf16* frx = (__bf16*)(smem + G_OFF_FRX) + wave * 4096;
  float*  stg = (float*) (smem + G_OFF_STG) + wave * (16 * 128);

  const unsigned lbase = (unsigned)(uintptr_t)smem;       // LDS byte offset base

  // ---------------- prologue: async weight/norm copies + TDM tile 0 --------
  if (!root) {
    for (int i = tid; i < (int)(WS_W12_SZ / 16); i += TPB)
      ASYNC_B128((unsigned)(lbase + G_OFF_W1 + i * 16), (unsigned)(i * 16), w12f);
  }
  {
    const float* enj = enormg + (size_t)jid * NE;
    for (int i = tid; i < 256; i += TPB)
      ASYNC_B128((unsigned)(lbase + G_OFF_EN + i * 16), (unsigned)(i * 16), enj);
  }
  if (tid == 0) tdm_load_1d(lbase + G_OFF_ET, embjt, 4096);   // tile 0 -> buf 0
  for (int i = tid; i < 256; i += TPB) { b1s[i] = b1[i]; lgs[i] = lng[i]; lbs[i] = lnb[i]; }
  if (tid < 128) b2s[tid] = b2[tid];
  WAIT_ASYNC0();
  __syncthreads();

  const int tok0 = blockIdx.x * TOK_BLK + wave * 16;

  if (!root) {
    // ---- stage x = [parent_e | z_j] straight into A-fragment layout ----
    for (int p = lane; p < 16 * 128; p += 32) {       // 128 bf16-pairs per row
      int m = p >> 7;
      int kp = (p & 127) * 2;
      int tg = tok0 + m;
      const float* src = (kp < ED) ? (zq + ((size_t)tg * NJ + pid) * ED + kp)
                                   : (z  + ((size_t)tg * NJ + jid) * ED + (kp - ED));
      *(unsigned*)&frx[fragpos16(m, kp)] = pack_bf16(src[0], src[1]);
    }
    __syncthreads();

    // ---- matmul1: h1 = x @ W1 ----
    v8f acc[16];
    #pragma unroll
    for (int nt = 0; nt < 16; ++nt) acc[nt] = (v8f){0,0,0,0,0,0,0,0};
    #pragma unroll
    for (int ks = 0; ks < 8; ++ks) {
      const v16bf a = *(const v16bf*)&frx[ks * 512 + lane * 16];
      #pragma unroll
      for (int nt = 0; nt < 16; ++nt) {
        const v16bf b = *(const v16bf*)&w1l[((ks * 16 + nt) * 32 + lane) * 16];
        acc[nt] = __builtin_amdgcn_wmma_f32_16x16x32_bf16(
            false, a, false, b, (short)0, acc[nt], false, false);
      }
    }

    // ---- +b1, LayerNorm stats ----
    float s1[8], s2[8];
    #pragma unroll
    for (int g = 0; g < 8; ++g) {
      float s = 0.0f, q = 0.0f;
      #pragma unroll
      for (int nt = 0; nt < 16; ++nt) {
        float v = acc[nt][g] + b1s[nt * 16 + col];
        acc[nt][g] = v;
        s += v; q = fmaf(v, v, q);
      }
      s1[g] = s; s2[g] = q;
    }
    #pragma unroll
    for (int off = 1; off < 16; off <<= 1) {
      #pragma unroll
      for (int g = 0; g < 8; ++g) {
        s1[g] += __shfl_xor(s1[g], off, 32);
        s2[g] += __shfl_xor(s2[g], off, 32);
      }
    }
    __syncthreads();          // x frags consumed; reuse frx for LN(h)
    #pragma unroll
    for (int g = 0; g < 8; ++g) {
      float mean = s1[g] * (1.0f / 256.0f);
      float var  = s2[g] * (1.0f / 256.0f) - mean * mean;
      float r    = rsqrtf(var + 1e-5f);
      int m = g + 8 * half;
      #pragma unroll
      for (int nt = 0; nt < 16; ++nt) {
        int n = nt * 16 + col;
        float v = (acc[nt][g] - mean) * r * lgs[n] + lbs[n];
        frx[fragpos16(m, n)] = (__bf16)(v > 0.0f ? v : 0.0f);   // ReLU, frag layout
      }
    }
    __syncthreads();

    // ---- matmul2: h2 = h @ W2 + b2 ----
    v8f acc2[8];
    #pragma unroll
    for (int nt = 0; nt < 8; ++nt) acc2[nt] = (v8f){0,0,0,0,0,0,0,0};
    #pragma unroll
    for (int ks = 0; ks < 8; ++ks) {
      const v16bf a = *(const v16bf*)&frx[ks * 512 + lane * 16];
      #pragma unroll
      for (int nt = 0; nt < 8; ++nt) {
        const v16bf b = *(const v16bf*)&w2l[((ks * 8 + nt) * 32 + lane) * 16];
        acc2[nt] = __builtin_amdgcn_wmma_f32_16x16x32_bf16(
            false, a, false, b, (short)0, acc2[nt], false, false);
      }
    }
    __syncthreads();          // h frags consumed; reuse frx for -2*h2
    #pragma unroll
    for (int g = 0; g < 8; ++g) {
      int m = g + 8 * half;
      #pragma unroll
      for (int nt = 0; nt < 8; ++nt) {
        int n = nt * 16 + col;
        float hv = acc2[nt][g] + b2s[n];
        stg[m * ED + n] = hv;                           // f32 for sqerr
        frx[fragpos16(m, n)] = (__bf16)(-2.0f * hv);    // A3 frag (exact -2x)
      }
    }
    __syncthreads();
  } else {
    // root joint: h = z
    for (int p = lane; p < 16 * 64; p += 32) {
      int m = p >> 6;
      int kp = (p & 63) * 2;
      const float* zz = z + ((size_t)(tok0 + m) * NJ + jid) * ED + kp;
      float v0 = zz[0], v1 = zz[1];
      stg[m * ED + kp] = v0; stg[m * ED + kp + 1] = v1;
      *(unsigned*)&frx[fragpos16(m, kp)] = pack_bf16(-2.0f * v0, -2.0f * v1);
    }
    __syncthreads();
  }

  // ---------------- nearest-code search: d = ||E||^2 - 2 h.E --------------
  v16bf a3[4];
  #pragma unroll
  for (int ks = 0; ks < 4; ++ks)
    a3[ks] = *(const v16bf*)&frx[ks * 512 + lane * 16];

  float bestv[8]; int besti[8];
  #pragma unroll
  for (int g = 0; g < 8; ++g) { bestv[g] = 3.4e38f; besti[g] = 0; }

  for (int t = 0; t < NE / 16; ++t) {
    if (tid == 0) WAIT_TENSOR0();           // tile t landed (wave0's TDM)
    __syncthreads();                        // visible to all; buf^1 free
    int cur = t & 1;
    if (tid == 0 && t + 1 < NE / 16)
      tdm_load_1d(lbase + G_OFF_ET + (cur ^ 1) * 4096,
                  (const char*)embjt + (size_t)(t + 1) * 4096, 4096);

    const __bf16* et = (const __bf16*)(smem + G_OFF_ET + cur * 4096);
    v8f d = (v8f){0,0,0,0,0,0,0,0};
    #pragma unroll
    for (int ks = 0; ks < 4; ++ks) {
      const v16bf b = *(const v16bf*)&et[(ks * 32 + lane) * 16];
      d = __builtin_amdgcn_wmma_f32_16x16x32_bf16(
          false, a3[ks], false, b, (short)0, d, false, false);
    }
    float nrm = enrm[t * 16 + col];
    int  nidx = t * 16 + col;
    #pragma unroll
    for (int g = 0; g < 8; ++g) {
      float dv = d[g] + nrm;
      if (dv < bestv[g]) { bestv[g] = dv; besti[g] = nidx; }
    }
  }

  // argmin reduce over the 16 lanes of each half
  #pragma unroll
  for (int off = 1; off < 16; off <<= 1) {
    #pragma unroll
    for (int g = 0; g < 8; ++g) {
      float ov = __shfl_xor(bestv[g], off, 32);
      int   oi = __shfl_xor(besti[g], off, 32);
      if (ov < bestv[g] || (ov == bestv[g] && oi < besti[g])) { bestv[g] = ov; besti[g] = oi; }
    }
  }

  // ---------------- write-out ----------------
  float errs = 0.0f;
  #pragma unroll
  for (int m = 0; m < 16; ++m) {
    const int g = m & 7;
    const int srcl = (m >> 3) << 4;
    int idx = __shfl(besti[g], srcl, 32);
    int tg = tok0 + m;
    const float* er = embj + (size_t)idx * ED;
    float* qo = zq + ((size_t)tg * NJ + jid) * ED;
    #pragma unroll
    for (int q = 0; q < 4; ++q) {
      int c = lane + q * 32;
      float ev = er[c];
      qo[c] = ev;                               // q == e numerically
      float dv = ev - stg[m * ED + c];
      errs = fmaf(dv, dv, errs);
    }
    if (lane == 0) oidx[(size_t)tg * NJ + jid] = (float)idx;
  }
  #pragma unroll
  for (int off = 1; off < 32; off <<= 1) errs += __shfl_xor(errs, off, 32);
  if (lane == 0) atomicAdd(loss, errs * LSCALE);
}

// ===========================================================================
// Fallback (self-contained, small-ws) level kernel — round-1 version
// ===========================================================================
#define OFF_W1   0
#define SZ_W1    (8*16*32*16*2)
#define OFF_W2   (OFF_W1 + SZ_W1)
#define SZ_W2    (8*8*32*16*2)
#define OFF_STG  (OFF_W2 + SZ_W2)
#define SZ_STG   (4*16*256*4)
#define OFF_ET   (OFF_STG + SZ_STG)
#define SZ_ET    (16*128*4)
#define OFF_EN   (OFF_ET + SZ_ET)
#define SZ_EN    (NE*4)
#define OFF_B1   (OFF_EN + SZ_EN)
#define OFF_LG   (OFF_B1 + 256*4)
#define OFF_LB   (OFF_LG + 256*4)
#define OFF_B2   (OFF_LB + 256*4)
#define LDS_TOTAL (OFF_B2 + 128*4)

__global__ __launch_bounds__(TPB, 1) void vq_level_fb(
    const float* __restrict__ z,   const float* __restrict__ emb,
    const float* __restrict__ W1,  const float* __restrict__ b1,
    const float* __restrict__ lng, const float* __restrict__ lnb,
    const float* __restrict__ W2,  const float* __restrict__ b2,
    float* __restrict__ zq, float* __restrict__ loss, float* __restrict__ oidx,
    Level lv)
{
  extern __shared__ __attribute__((aligned(64))) char smem[];
  __bf16* w1f  = (__bf16*)(smem + OFF_W1);
  __bf16* w2f  = (__bf16*)(smem + OFF_W2);
  float*  stga = (float*) (smem + OFF_STG);
  float*  etile= (float*) (smem + OFF_ET);
  float*  enrm = (float*) (smem + OFF_EN);
  float*  b1s  = (float*) (smem + OFF_B1);
  float*  lgs  = (float*) (smem + OFF_LG);
  float*  lbs  = (float*) (smem + OFF_LB);
  float*  b2s  = (float*) (smem + OFF_B2);

  const int tid = threadIdx.x, lane = tid & 31, wave = tid >> 5;
  const int half = lane >> 4, col = lane & 15;
  const int jid = lv.j[blockIdx.y], pid = lv.p[blockIdx.y];
  const bool root = (pid < 0);
  const float* embj = emb + (size_t)jid * NE * ED;
  float* stg = stga + wave * 16 * 256;

  if (!root) {
    for (int f = tid; f < 8*16*32*16; f += TPB) {
      int e = f & 15, ln = (f>>4)&31, nt = (f>>9)&15, ks = f>>13;
      w1f[f] = (__bf16)W1[(ks*32 + kmap16(e, ln>>4)) * HIDN + nt*16 + (ln&15)];
    }
    for (int f = tid; f < 8*8*32*16; f += TPB) {
      int e = f & 15, ln = (f>>4)&31, nt = (f>>9)&7, ks = f>>12;
      w2f[f] = (__bf16)W2[(ks*32 + kmap16(e, ln>>4)) * ED + nt*16 + (ln&15)];
    }
  }
  for (int c = tid; c < NE; c += TPB) {
    const float* r = embj + (size_t)c * ED;
    float s = 0.0f;
    #pragma unroll 8
    for (int k = 0; k < ED; ++k) s = fmaf(r[k], r[k], s);
    enrm[c] = s;
  }
  for (int i = tid; i < 256; i += TPB) { b1s[i]=b1[i]; lgs[i]=lng[i]; lbs[i]=lnb[i]; }
  if (tid < 128) b2s[tid] = b2[tid];
  __syncthreads();

  const int tok0 = blockIdx.x * TOK_BLK + wave * 16;

  if (!root) {
    for (int m = 0; m < 16; ++m) {
      int tg = tok0 + m;
      const float* pe = zq + ((size_t)tg*NJ + pid)*ED;
      const float* zz = z  + ((size_t)tg*NJ + jid)*ED;
      for (int c = lane; c < ED; c += 32) { stg[m*256+c]=pe[c]; stg[m*256+ED+c]=zz[c]; }
    }
    __syncthreads();
    v8f acc[16];
    #pragma unroll
    for (int nt = 0; nt < 16; ++nt) acc[nt] = (v8f){0,0,0,0,0,0,0,0};
    #pragma unroll
    for (int ks = 0; ks < 8; ++ks) {
      v16bf a;
      #pragma unroll
      for (int e = 0; e < 16; ++e) a[e] = (__bf16)stg[col*256 + ks*32 + kmap16(e, half)];
      #pragma unroll
      for (int nt = 0; nt < 16; ++nt) {
        const v16bf b = *(const v16bf*)&w1f[((ks*16+nt)*32 + lane)*16];
        acc[nt] = __builtin_amdgcn_wmma_f32_16x16x32_bf16(false,a,false,b,(short)0,acc[nt],false,false);
      }
    }
    float s1[8], s2[8];
    #pragma unroll
    for (int g = 0; g < 8; ++g) {
      float s=0.f, q=0.f;
      #pragma unroll
      for (int nt = 0; nt < 16; ++nt) {
        float v = acc[nt][g] + b1s[nt*16+col]; acc[nt][g]=v; s+=v; q=fmaf(v,v,q);
      }
      s1[g]=s; s2[g]=q;
    }
    #pragma unroll
    for (int off = 1; off < 16; off <<= 1) {
      #pragma unroll
      for (int g = 0; g < 8; ++g) { s1[g]+=__shfl_xor(s1[g],off,32); s2[g]+=__shfl_xor(s2[g],off,32); }
    }
    __syncthreads();
    #pragma unroll
    for (int g = 0; g < 8; ++g) {
      float mean=s1[g]*(1.f/256.f), var=s2[g]*(1.f/256.f)-mean*mean, r=rsqrtf(var+1e-5f);
      int m = g + 8*half;
      #pragma unroll
      for (int nt = 0; nt < 16; ++nt) {
        int n = nt*16+col;
        float v = (acc[nt][g]-mean)*r*lgs[n]+lbs[n];
        stg[m*256+n] = v > 0.f ? v : 0.f;
      }
    }
    __syncthreads();
    v8f acc2[8];
    #pragma unroll
    for (int nt = 0; nt < 8; ++nt) acc2[nt] = (v8f){0,0,0,0,0,0,0,0};
    #pragma unroll
    for (int ks = 0; ks < 8; ++ks) {
      v16bf a;
      #pragma unroll
      for (int e = 0; e < 16; ++e) a[e] = (__bf16)stg[col*256 + ks*32 + kmap16(e, half)];
      #pragma unroll
      for (int nt = 0; nt < 8; ++nt) {
        const v16bf b = *(const v16bf*)&w2f[((ks*8+nt)*32 + lane)*16];
        acc2[nt] = __builtin_amdgcn_wmma_f32_16x16x32_bf16(false,a,false,b,(short)0,acc2[nt],false,false);
      }
    }
    __syncthreads();
    #pragma unroll
    for (int g = 0; g < 8; ++g) {
      int m = g + 8*half;
      #pragma unroll
      for (int nt = 0; nt < 8; ++nt) stg[m*256 + nt*16+col] = acc2[nt][g] + b2s[nt*16+col];
    }
    __syncthreads();
  } else {
    for (int m = 0; m < 16; ++m) {
      const float* zz = z + ((size_t)(tok0+m)*NJ + jid)*ED;
      for (int c = lane; c < ED; c += 32) stg[m*256+c] = zz[c];
    }
    __syncthreads();
  }

  v16bf a3[4];
  #pragma unroll
  for (int ks = 0; ks < 4; ++ks) {
    #pragma unroll
    for (int e = 0; e < 16; ++e)
      a3[ks][e] = (__bf16)(-2.0f * stg[col*256 + ks*32 + kmap16(e, half)]);
  }
  float bestv[8]; int besti[8];
  #pragma unroll
  for (int g = 0; g < 8; ++g) { bestv[g]=3.4e38f; besti[g]=0; }
  for (int t = 0; t < NE/16; ++t) {
    __syncthreads();
    for (int i = tid; i < 16*ED; i += TPB) etile[i] = embj[(size_t)t*16*ED + i];
    __syncthreads();
    v8f d = (v8f){0,0,0,0,0,0,0,0};
    #pragma unroll
    for (int ks = 0; ks < 4; ++ks) {
      v16bf b;
      #pragma unroll
      for (int e = 0; e < 16; ++e) b[e] = (__bf16)etile[col*ED + ks*32 + kmap16(e, half)];
      d = __builtin_amdgcn_wmma_f32_16x16x32_bf16(false,a3[ks],false,b,(short)0,d,false,false);
    }
    float nrm = enrm[t*16+col]; int nidx = t*16+col;
    #pragma unroll
    for (int g = 0; g < 8; ++g) {
      float dv = d[g] + nrm;
      if (dv < bestv[g]) { bestv[g]=dv; besti[g]=nidx; }
    }
  }
  #pragma unroll
  for (int off = 1; off < 16; off <<= 1) {
    #pragma unroll
    for (int g = 0; g < 8; ++g) {
      float ov=__shfl_xor(bestv[g],off,32); int oi=__shfl_xor(besti[g],off,32);
      if (ov<bestv[g] || (ov==bestv[g] && oi<besti[g])) { bestv[g]=ov; besti[g]=oi; }
    }
  }
  float errs = 0.0f;
  #pragma unroll
  for (int m = 0; m < 16; ++m) {
    const int g = m & 7, srcl = (m >> 3) << 4;
    int idx = __shfl(besti[g], srcl, 32);
    int tg = tok0 + m;
    const float* er = embj + (size_t)idx * ED;
    float* qo = zq + ((size_t)tg*NJ + jid)*ED;
    #pragma unroll
    for (int q = 0; q < 4; ++q) {
      int c = lane + q*32;
      float ev = er[c]; qo[c] = ev;
      float dv = ev - stg[m*256 + c];
      errs = fmaf(dv, dv, errs);
    }
    if (lane == 0) oidx[(size_t)tg*NJ + jid] = (float)idx;
  }
  #pragma unroll
  for (int off = 1; off < 32; off <<= 1) errs += __shfl_xor(errs, off, 32);
  if (lane == 0) atomicAdd(loss, errs * LSCALE);
}

// ===========================================================================
// Host side
// ===========================================================================
static const int LV_N[11]    = {1, 3, 3, 5, 5, 5, 2, 2, 2, 2, 2};
static const int LV_J[11][5] = {
  {0},           {1, 6, 11},    {2, 7, 12},
  {3, 8, 13, 16, 24}, {4, 9, 14, 17, 25}, {5, 10, 15, 18, 26},
  {19, 27}, {20, 28}, {21, 29}, {22, 30}, {23, 31}};
static const int LV_P[11][5] = {
  {-1},          {0, 0, 0},     {1, 6, 11},
  {2, 7, 12, 12, 12}, {3, 8, 13, 16, 24}, {4, 9, 14, 17, 25},
  {18, 26}, {19, 27}, {20, 28}, {21, 29}, {22, 30}};

extern "C" void kernel_launch(void* const* d_in, const int* in_sizes, int n_in,
                              void* d_out, int out_size, void* d_ws, size_t ws_size,
                              hipStream_t stream) {
  (void)in_sizes; (void)n_in; (void)out_size;
  const float* z   = (const float*)d_in[0];
  const float* emb = (const float*)d_in[1];
  const float* W1  = (const float*)d_in[2];
  const float* b1  = (const float*)d_in[3];
  const float* lng = (const float*)d_in[4];
  const float* lnb = (const float*)d_in[5];
  const float* W2  = (const float*)d_in[6];
  const float* b2  = (const float*)d_in[7];

  float* out  = (float*)d_out;
  float* zq   = out;
  float* loss = out + (size_t)TOKENS * NJ * ED;
  float* oidx = loss + 1;

  vq_init_kernel<<<1, 1, 0, stream>>>(loss);

  const bool fast = (d_ws != nullptr) && (ws_size >= WS_NEED);
  if (fast) {
    __bf16* embf  = (__bf16*)((char*)d_ws + WS_EMBF);
    __bf16* w12f  = (__bf16*)((char*)d_ws + WS_W12);
    float*  enorm = (float*) ((char*)d_ws + WS_EN);

    (void)hipFuncSetAttribute((const void*)vq_level_fast,
                              hipFuncAttributeMaxDynamicSharedMemorySize, G_LDS_TOTAL);
    prep_w_kernel    <<<384, 256, 0, stream>>>(W1, W2, w12f);
    prep_emb_kernel  <<<dim3(64, 32), TPB, 0, stream>>>(emb, embf);
    prep_enorm_kernel<<<256, TPB, 0, stream>>>(emb, enorm);

    for (int L = 0; L < 11; ++L) {
      Level lv{};
      lv.n = LV_N[L];
      for (int i = 0; i < lv.n; ++i) { lv.j[i] = LV_J[L][i]; lv.p[i] = LV_P[L][i]; }
      dim3 grid(TOKENS / TOK_BLK, lv.n);
      vq_level_fast<<<grid, TPB, G_LDS_TOTAL, stream>>>(
          z, emb, embf, w12f, enorm, b1, lng, lnb, b2, zq, loss, oidx, lv);
    }
  } else {
    (void)hipFuncSetAttribute((const void*)vq_level_fb,
                              hipFuncAttributeMaxDynamicSharedMemorySize, LDS_TOTAL);
    for (int L = 0; L < 11; ++L) {
      Level lv{};
      lv.n = LV_N[L];
      for (int i = 0; i < lv.n; ++i) { lv.j[i] = LV_J[L][i]; lv.p[i] = LV_P[L][i]; }
      dim3 grid(TOKENS / TOK_BLK, lv.n);
      vq_level_fb<<<grid, TPB, LDS_TOTAL, stream>>>(
          z, emb, W1, b1, lng, lnb, W2, b2, zq, loss, oidx, lv);
    }
  }
}
